// EmbedMatcher_30030411334232
// MI455X (gfx1250) — compile-verified, matrix operand order checked by reading, not run on hardware
//
#include <hip/hip_runtime.h>
#include <math.h>

// ---------------- WMMA types / helpers (gfx1250, wave32) ----------------
typedef __attribute__((ext_vector_type(16))) __bf16 v16bf;
typedef __attribute__((ext_vector_type(8)))  float  v8f;
typedef unsigned short ushort16;

union Frag { v16bf v; uint4 q[2]; };

__device__ __forceinline__ unsigned short pack1_bf16(float f) {
  union { float f; unsigned int u; } v; v.f = f;
  unsigned int u = v.u + 0x7FFFu + ((v.u >> 16) & 1u);   // RNE
  return (unsigned short)(u >> 16);
}
__device__ __forceinline__ unsigned int pack2_bf16(float lo, float hi) {
  union { float f; unsigned int u; } a, b;
  a.f = lo; b.f = hi;
  unsigned int ua = a.u + 0x7FFFu + ((a.u >> 16) & 1u);
  unsigned int ub = b.u + 0x7FFFu + ((b.u >> 16) & 1u);
  return (ua >> 16) | (ub & 0xFFFF0000u);
}
__device__ __forceinline__ float sigf(float x) { return 1.0f / (1.0f + expf(-x)); }

// ---------------- WMMA GEMM on pre-converted bf16 ----------------
// C[M,N] = A[M,K](bf16) * W[N,K](bf16)^T (+bias, +f32 resid, relu)
// Wave tile: 32 rows x 64 cols = 2 A-frags x 4 B-frags = 8 WMMAs / K-step.
// Fragments load as 2x b128 per lane straight from row-major bf16 (no packing VALU).
__global__ void __launch_bounds__(128)
gemm_wmma_bf16(const unsigned short* __restrict__ A, int lda,
               const unsigned short* __restrict__ W, int ldw,
               float* __restrict__ Cf, unsigned short* __restrict__ Cb, int ldc,
               const float* __restrict__ addvec,
               const float* __restrict__ resid, int ldr,
               int M, int N, int K, int relu)
{
  const int lane = threadIdx.x & 31;
  const int wid  = threadIdx.x >> 5;
  const int ntn  = N >> 6;                         // 64-wide tiles along N
  const int tile = blockIdx.x * 4 + wid;
  const int m0   = (tile / ntn) * 32;
  const int n0   = (tile % ntn) * 64;
  if (m0 >= M) return;

  const int l16  = lane & 15;
  const int half = lane >> 4;

  v8f acc[2][4];
#pragma unroll
  for (int mi = 0; mi < 2; ++mi)
#pragma unroll
    for (int s = 0; s < 4; ++s)
#pragma unroll
      for (int r = 0; r < 8; ++r) acc[mi][s][r] = 0.0f;

  const unsigned short* a0 = A + (size_t)(m0 + l16) * lda + 8 * half;
  const unsigned short* a1 = a0 + (size_t)16 * lda;
  const unsigned short* w0 = W + (size_t)(n0 + l16) * ldw + 8 * half;

  for (int kk = 0; kk < K; kk += 32) {
    Frag fa0, fa1;
    fa0.q[0] = *(const uint4*)(a0 + kk);
    fa0.q[1] = *(const uint4*)(a0 + kk + 16);
    fa1.q[0] = *(const uint4*)(a1 + kk);
    fa1.q[1] = *(const uint4*)(a1 + kk + 16);
#pragma unroll
    for (int s = 0; s < 4; ++s) {
      Frag fb;
      const unsigned short* wp = w0 + (size_t)(s * 16) * ldw + kk;
      fb.q[0] = *(const uint4*)(wp);
      fb.q[1] = *(const uint4*)(wp + 16);
      acc[0][s] = __builtin_amdgcn_wmma_f32_16x16x32_bf16(
          false, fa0.v, false, fb.v, (short)0, acc[0][s], false, false);
      acc[1][s] = __builtin_amdgcn_wmma_f32_16x16x32_bf16(
          false, fa1.v, false, fb.v, (short)0, acc[1][s], false, false);
    }
  }

  // D layout: lanes 0-15: M=r, N=lane; lanes 16-31: M=8+r, N=lane-16.
#pragma unroll
  for (int mi = 0; mi < 2; ++mi) {
#pragma unroll
    for (int s = 0; s < 4; ++s) {
      const int n = n0 + s * 16 + l16;
#pragma unroll
      for (int r = 0; r < 8; ++r) {
        const int m = m0 + 16 * mi + r + 8 * half;
        float v = acc[mi][s][r];
        if (addvec) v += addvec[n];
        if (resid)  v += resid[(size_t)m * ldr + n];
        if (relu)   v = fmaxf(v, 0.0f);
        if (Cf) Cf[(size_t)m * ldc + n] = v;
        if (Cb) Cb[(size_t)m * ldc + n] = pack1_bf16(v);
      }
    }
  }
}

// ---------------- f32 -> bf16 bulk convert (2 elems/thread, dword stores) ---------
__global__ void conv_bf16_kernel(const float* __restrict__ in, unsigned int* __restrict__ out) {
  const int i = blockIdx.x * 256 + threadIdx.x;   // over n/2
  out[i] = pack2_bf16(in[2 * i], in[2 * i + 1]);
}

// ---------------- Neighbor encoder: one block (128 thr) per row ----------------
#define DD   128
#define NNB  64
#define KNN  10

__global__ void __launch_bounds__(128)
ne_kernel(const int* __restrict__ conn, const int* __restrict__ ids, int id_col,
          const float* __restrict__ emb, const float* __restrict__ gcn_W,
          const float* __restrict__ gcn_wb, const float* __restrict__ gcn_b,
          float* __restrict__ out, unsigned short* __restrict__ out_bf, int out_col)
{
  __shared__ __align__(16) float s_center[DD];
  __shared__ __align__(16) float s_cat[2 * DD];
  __shared__ float s_sim[NNB];
  __shared__ int   s_sel[KNN];
  __shared__ float s_cnorm;

  const int b = blockIdx.x;
  const int t = threadIdx.x;
  const int cid = ids[b * 2 + id_col];
  s_center[t] = emb[(size_t)cid * DD + t];
  __syncthreads();

  if (t < 32) {
    float p = 0.f;
    for (int j = t; j < DD; j += 32) p += s_center[j] * s_center[j];
    for (int o = 16; o; o >>= 1) p += __shfl_xor(p, o, 32);
    if (t == 0) s_cnorm = p;
  }
  __syncthreads();

  if (t < NNB) {
    const int eid = conn[((size_t)b * NNB + t) * 2 + 1];
    const float4* ep = (const float4*)(emb + (size_t)eid * DD);
    float dot = 0.f, nn2 = 0.f;
    for (int j = 0; j < DD / 4; ++j) {
      float4 e = ep[j];
      float4 c = *(const float4*)(s_center + 4 * j);
      dot += e.x * c.x + e.y * c.y + e.z * c.z + e.w * c.w;
      nn2 += e.x * e.x + e.y * e.y + e.z * e.z + e.w * e.w;
    }
    float den = fmaxf(sqrtf(nn2), 1e-8f) * fmaxf(sqrtf(s_cnorm), 1e-8f);
    s_sim[t] = dot / den;
  }
  __syncthreads();

  if (t == 0) {                                   // top-10, ties -> lowest index
    for (int k = 0; k < KNN; ++k) {
      int best = 0; float bv = s_sim[0];
      for (int n = 1; n < NNB; ++n)
        if (s_sim[n] > bv) { bv = s_sim[n]; best = n; }
      s_sel[k] = best;
      s_sim[best] = -3.0e38f;
    }
  }
  __syncthreads();

  {
    float racc = 0.f, eacc = 0.f;
    for (int k = 0; k < KNN; ++k) {
      const int i   = s_sel[k];
      const int rid = conn[((size_t)b * NNB + i) * 2 + 0];
      const int eid = conn[((size_t)b * NNB + i) * 2 + 1];
      racc += emb[(size_t)rid * DD + t];
      eacc += emb[(size_t)eid * DD + t];
    }
    s_cat[t]      = racc * (1.0f / KNN);
    s_cat[DD + t] = eacc * (1.0f / KNN);
  }
  __syncthreads();

  {
    const float4* wp = (const float4*)(gcn_W + (size_t)t * 2 * DD);
    float acc = 0.f;
    for (int j = 0; j < (2 * DD) / 4; ++j) {
      float4 w = wp[j];
      float4 c = *(const float4*)(s_cat + 4 * j);
      acc += w.x * c.x + w.y * c.y + w.z * c.z + w.w * c.w;
    }
    const float v = tanhf(acc + gcn_wb[t] + gcn_b[t]);
    out   [(size_t)b * (2 * DD) + out_col + t] = v;
    out_bf[(size_t)b * (2 * DD) + out_col + t] = pack1_bf16(v);
  }
}

// ---------------- LayerNorm (256 dims, wave per row); optional bf16 mirror -------
__global__ void __launch_bounds__(128)
ln_kernel(const float* __restrict__ y, const float* __restrict__ g,
          const float* __restrict__ b,
          float* __restrict__ dst_f, int f_stride,
          unsigned short* __restrict__ dst_bf, int bf_stride)
{
  const int lane = threadIdx.x & 31;
  const int row  = blockIdx.x * 4 + (threadIdx.x >> 5);
  const float* yr = y + (size_t)row * 256;
  float v[8];
  float s = 0.f;
#pragma unroll
  for (int j = 0; j < 8; ++j) { v[j] = yr[lane + 32 * j]; s += v[j]; }
  for (int o = 16; o; o >>= 1) s += __shfl_xor(s, o, 32);
  const float mu = s * (1.0f / 256.0f);
  float q = 0.f;
#pragma unroll
  for (int j = 0; j < 8; ++j) { float d = v[j] - mu; q += d * d; }
  for (int o = 16; o; o >>= 1) q += __shfl_xor(q, o, 32);
  const float inv = rsqrtf(q * (1.0f / 256.0f) + 1e-5f);
#pragma unroll
  for (int j = 0; j < 8; ++j) {
    const int c = lane + 32 * j;
    const float o2 = (v[j] - mu) * inv * g[c] + b[c];
    dst_f[(size_t)row * f_stride + c] = o2;
    if (dst_bf) dst_bf[(size_t)row * bf_stride + c] = pack1_bf16(o2);
  }
}

// ---------------- support_g = mean over 128 support rows ----------------
__global__ void mean_kernel(const float* __restrict__ y, float* __restrict__ sg) {
  const int d = threadIdx.x;                      // 256 threads
  float s = 0.f;
  for (int r = 0; r < 128; ++r) s += y[r * 256 + d];
  sg[d] = s * (1.0f / 128.0f);
}

// ---------------- Wcat(bf16)[n,0:256]=W_ih[n,:]; [n,256:512]=W_hh[n,0:256] --------
__global__ void wcat_kernel(const float* __restrict__ W_ih, const float* __restrict__ W_hh,
                            unsigned short* __restrict__ Wcat) {
  const int idx = blockIdx.x * 256 + threadIdx.x; // 2048*512 total
  const int n = idx >> 9, k = idx & 511;
  const float v = (k < 256) ? W_ih[n * 256 + k] : W_hh[n * 512 + (k - 256)];
  Wcat[idx] = pack1_bf16(v);
}

// ---------------- bias vectors: bvec=b_ih+b_hh; bvec2=bvec + sg @ W_hh[:,256:]^T --
__global__ void sgt_kernel(const float* __restrict__ sg, const float* __restrict__ W_hh,
                           const float* __restrict__ b_ih, const float* __restrict__ b_hh,
                           float* __restrict__ bvec, float* __restrict__ bvec2) {
  const int n = blockIdx.x * 256 + threadIdx.x;   // 2048
  float s = 0.f;
  const float* w = W_hh + (size_t)n * 512 + 256;
  for (int k = 0; k < 256; ++k) s += sg[k] * w[k];
  const float bb = b_ih[n] + b_hh[n];
  bvec[n]  = bb;
  bvec2[n] = bb + s;
}

// ---------------- zero upper (h) half of X_bf ----------------
__global__ void xinit_kernel(unsigned int* __restrict__ Xbf) {
  const int idx = blockIdx.x * 256 + threadIdx.x; // 4096*128 dwords
  const int r = idx >> 7, c = idx & 127;
  Xbf[(size_t)r * 256 + 128 + c] = 0u;            // row = 512 bf16 = 256 dwords
}

// ------- LSTM cell elementwise; h -> h_f32 and X_bf[:,256:512] (bf16) -------
__global__ void lstm_ewise(const float* __restrict__ gates, float* __restrict__ c,
                           const float* __restrict__ qg, float* __restrict__ h,
                           unsigned short* __restrict__ Xbf, int first) {
  const int idx = blockIdx.x * 256 + threadIdx.x; // 4096*512
  const int r = idx >> 9, j = idx & 511;
  const float* gr = gates + (size_t)r * 2048;
  const float gi = gr[j], gf = gr[512 + j], gg = gr[1024 + j], go = gr[1536 + j];
  const float cp = first ? 0.0f : c[idx];
  const float cn = sigf(gf) * cp + sigf(gi) * tanhf(gg);
  c[idx] = cn;
  if (j < 256) {
    const float hv = qg[(size_t)r * 256 + j] + sigf(go) * tanhf(cn);
    h[(size_t)r * 256 + j] = hv;
    Xbf[(size_t)r * 512 + 256 + j] = pack1_bf16(hv);
  }
}

// ---------------- final cosine: wave per row ----------------
__global__ void __launch_bounds__(128)
cos_kernel(const float* __restrict__ h, const float* __restrict__ sg, float* __restrict__ out) {
  const int lane = threadIdx.x & 31;
  const int b = blockIdx.x * 4 + (threadIdx.x >> 5);
  const float* q = h + (size_t)b * 256;
  float dot = 0.f, qn = 0.f, sn = 0.f;
#pragma unroll
  for (int j = 0; j < 8; ++j) {
    const int c = lane + 32 * j;
    const float qv = q[c], sv = sg[c];
    dot += qv * sv; qn += qv * qv; sn += sv * sv;
  }
  for (int o = 16; o; o >>= 1) {
    dot += __shfl_xor(dot, o, 32);
    qn  += __shfl_xor(qn,  o, 32);
    sn  += __shfl_xor(sn,  o, 32);
  }
  if (lane == 0)
    out[b] = dot / (fmaxf(sqrtf(qn), 1e-12f) * fmaxf(sqrtf(sn), 1e-12f));
}

// =====================================================================================
extern "C" void kernel_launch(void* const* d_in, const int* in_sizes, int n_in,
                              void* d_out, int out_size, void* d_ws, size_t ws_size,
                              hipStream_t stream) {
  const int*   query    = (const int*)d_in[0];
  const int*   support  = (const int*)d_in[1];
  const int*   q_l_conn = (const int*)d_in[2];
  const int*   q_r_conn = (const int*)d_in[4];
  const int*   s_l_conn = (const int*)d_in[6];
  const int*   s_r_conn = (const int*)d_in[8];
  const float* emb      = (const float*)d_in[10];
  const float* gcn_W    = (const float*)d_in[11];
  const float* gcn_wb   = (const float*)d_in[12];
  const float* gcn_b    = (const float*)d_in[13];
  const float* se_w1    = (const float*)d_in[14];
  const float* se_b1    = (const float*)d_in[15];
  const float* se_w2    = (const float*)d_in[16];
  const float* se_b2    = (const float*)d_in[17];
  const float* ln_g     = (const float*)d_in[18];
  const float* ln_b     = (const float*)d_in[19];
  const float* W_ih     = (const float*)d_in[20];
  const float* W_hh     = (const float*)d_in[21];
  const float* b_ih     = (const float*)d_in[22];
  const float* b_hh     = (const float*)d_in[23];

  float* p = (float*)d_ws;
  float* q_nb    = p; p += 4096 * 256;            // f32 (residual)
  float* s_nb    = p; p += 128  * 256;
  float* y_q     = p; p += 4096 * 256;            // pre-LN
  float* y_s     = p; p += 128  * 256;
  float* qg      = p; p += 4096 * 256;            // query_g f32
  float* hbuf    = p; p += 4096 * 256;            // LSTM h f32
  float* sg      = p; p += 256;
  float* bvec    = p; p += 2048;
  float* bvec2   = p; p += 2048;
  float* gates   = p; p += 4096 * 2048;
  float* cst     = p; p += 4096 * 512;
  unsigned short* q_nb_bf   = (unsigned short*)p; p += 4096 * 256 / 2;
  unsigned short* s_nb_bf   = (unsigned short*)p; p += 128  * 256 / 2;
  unsigned short* w1_bf     = (unsigned short*)p; p += 512  * 256 / 2;
  unsigned short* w2_bf     = (unsigned short*)p; p += 256  * 512 / 2;
  unsigned short* Wcat_bf   = (unsigned short*)p; p += 2048 * 512 / 2;
  unsigned short* hmid_q_bf = (unsigned short*)p; p += 4096 * 512 / 2;
  unsigned short* hmid_s_bf = (unsigned short*)p; p += 128  * 512 / 2;
  unsigned short* X_bf      = (unsigned short*)p; p += 4096 * 512 / 2;  // [query_g | h] bf16

  // 0) one-time weight conversions / packing
  conv_bf16_kernel<<<(512 * 256 / 2) / 256, 256, 0, stream>>>(se_w1, (unsigned int*)w1_bf);
  conv_bf16_kernel<<<(256 * 512 / 2) / 256, 256, 0, stream>>>(se_w2, (unsigned int*)w2_bf);
  wcat_kernel<<<(2048 * 512) / 256, 256, 0, stream>>>(W_ih, W_hh, Wcat_bf);

  // 1) neighbor encoders (emit f32 + bf16)
  ne_kernel<<<4096, 128, 0, stream>>>(q_l_conn, query,   0, emb, gcn_W, gcn_wb, gcn_b, q_nb, q_nb_bf, 0);
  ne_kernel<<<4096, 128, 0, stream>>>(q_r_conn, query,   1, emb, gcn_W, gcn_wb, gcn_b, q_nb, q_nb_bf, 128);
  ne_kernel<<<128,  128, 0, stream>>>(s_l_conn, support, 0, emb, gcn_W, gcn_wb, gcn_b, s_nb, s_nb_bf, 0);
  ne_kernel<<<128,  128, 0, stream>>>(s_r_conn, support, 1, emb, gcn_W, gcn_wb, gcn_b, s_nb, s_nb_bf, 128);

  // 2) support encoder FFN + residual + LN; support branch -> mean -> support_g
  gemm_wmma_bf16<<<8, 128, 0, stream>>>(s_nb_bf, 256, w1_bf, 256, nullptr, hmid_s_bf, 512,
                                        se_b1, nullptr, 0, 128, 512, 256, 1);
  gemm_wmma_bf16<<<4, 128, 0, stream>>>(hmid_s_bf, 512, w2_bf, 512, y_s, nullptr, 256,
                                        se_b2, s_nb, 256, 128, 256, 512, 0);
  ln_kernel<<<32, 128, 0, stream>>>(y_s, ln_g, ln_b, y_s, 256, nullptr, 0);
  mean_kernel<<<1, 256, 0, stream>>>(y_s, sg);

  // query branch; LN emits qg (f32) + X_bf[:, :256] (bf16)
  gemm_wmma_bf16<<<256, 128, 0, stream>>>(q_nb_bf, 256, w1_bf, 256, nullptr, hmid_q_bf, 512,
                                          se_b1, nullptr, 0, 4096, 512, 256, 1);
  gemm_wmma_bf16<<<128, 128, 0, stream>>>(hmid_q_bf, 512, w2_bf, 512, y_q, nullptr, 256,
                                          se_b2, q_nb, 256, 4096, 256, 512, 0);
  ln_kernel<<<1024, 128, 0, stream>>>(y_q, ln_g, ln_b, qg, 256, X_bf, 512);
  xinit_kernel<<<(4096 * 128) / 256, 256, 0, stream>>>((unsigned int*)X_bf);

  // 3) LSTM prep (softmax([B,1])==1 => r = support_g, folded into bvec2)
  sgt_kernel<<<8, 256, 0, stream>>>(sg, W_hh, b_ih, b_hh, bvec, bvec2);

  // 4) 4 process steps: gates = X @ Wcat^T + bias(+sgt for steps>=2)
  for (int t = 0; t < 4; ++t) {
    gemm_wmma_bf16<<<1024, 128, 0, stream>>>(X_bf, 512, Wcat_bf, 512, gates, nullptr, 2048,
                                             (t == 0) ? bvec : bvec2, nullptr, 0,
                                             4096, 2048, 512, 0);
    lstm_ewise<<<8192, 256, 0, stream>>>(gates, cst, qg, hbuf, X_bf, (t == 0) ? 1 : 0);
  }

  // 5) cosine(qf, support_g)
  cos_kernel<<<1024, 128, 0, stream>>>(hbuf, sg, (float*)d_out);
}
